// Attention_27771258536423
// MI455X (gfx1250) — compile-verified, matrix operand order checked by reading, not run on hardware
//
#include <hip/hip_runtime.h>
#include <hip/hip_bf16.h>

// Problem constants
#define DIM   768
#define NH    12
#define HD    64
#define NTOK  197
#define NPAD  224          // 14*16, 7*32
#define BATCH 64
#define MROWS (BATCH * NTOK)   // 12608
#define MTILES (MROWS / 16)    // 788 (exact)
#define QSCALE 0.125f          // 64^-0.5

typedef __attribute__((ext_vector_type(16))) __bf16        v16bf;
typedef __attribute__((ext_vector_type(8)))  float         v8f;
typedef __attribute__((ext_vector_type(4)))  unsigned int  uint4v;

union FragU { v16bf bf; uint4v u[2]; };

// A-frag / B-frag loader for V_WMMA_F32_16X16X32_BF16.
// Per ISA 7.12.2: lane holds row (lane&15); halves 0..7 at K = kbase+koff,
// halves 8..15 at K = kbase+koff+16, koff = (lane&16)?8:0.
// Both chunks are contiguous 16B -> two b128 loads (global or ds).
__device__ __forceinline__ v16bf load_frag(const __hip_bfloat16* base, int ld,
                                           int kbase, int lane) {
  const __hip_bfloat16* p =
      base + (size_t)(lane & 15) * ld + kbase + ((lane & 16) ? 8 : 0);
  FragU f;
  f.u[0] = *(const uint4v*)(p);
  f.u[1] = *(const uint4v*)(p + 16);
  return f.bf;
}

__device__ __forceinline__ v8f wmma_bf16(v16bf a, v16bf b, v8f c) {
  return __builtin_amdgcn_wmma_f32_16x16x32_bf16(false, a, false, b,
                                                 (short)0, c, false, false);
}

// Async global->LDS copy of one 16B chunk (GV mode, SADDR=off).
// lds_off is the byte offset within the workgroup LDS allocation (each kernel
// has a single __shared__ block, allocated at offset 0). lds_obj is the base
// pointer of that shared block: it is unused in the template but escapes the
// object into the asm so that, combined with the "memory" clobber, the
// compiler must assume the asm writes the LDS array (otherwise it folds the
// later ds_load fragments to undef -- observed in round 3).
__device__ __forceinline__ void async_load_b128(unsigned lds_off,
                                                const void* gptr,
                                                void* lds_obj) {
  asm volatile("global_load_async_to_lds_b128 %0, %1, off"
               :: "v"(lds_off), "v"(gptr), "v"(lds_obj)
               : "memory");
}

__device__ __forceinline__ void wait_asynccnt0() {
  asm volatile("s_wait_asynccnt 0x0" ::: "memory");
}

// ---------------- fp32 -> bf16 convert / zero ----------------
__global__ void k_convert(const float* __restrict__ in,
                          __hip_bfloat16* __restrict__ out, int n) {
  int i = blockIdx.x * blockDim.x + threadIdx.x;
  if (i < n) out[i] = __float2bfloat16(in[i]);
}

__global__ void k_zero(unsigned int* __restrict__ p, int nwords) {
  int i = blockIdx.x * blockDim.x + threadIdx.x;
  if (i < nwords) p[i] = 0u;
}

// ---------------- QKV GEMM: [12608,768]bf16 @ W^T[768,2304] ----------------
// Block: 8 waves. A tile (16x768, 24.5KB) async-staged to LDS once and shared
// by all waves; each wave computes a 16x32 tile streaming B from global.
// Epilogue scatters into Q*scale [BH,NPAD,HD], K [BH,NPAD,HD], V^T [BH,HD,NPAD].
__global__ void __launch_bounds__(256) k_qkv(
    const __hip_bfloat16* __restrict__ xb, const __hip_bfloat16* __restrict__ wb,
    const float* __restrict__ q_bias, const float* __restrict__ v_bias,
    __hip_bfloat16* __restrict__ qs, __hip_bfloat16* __restrict__ kmat,
    __hip_bfloat16* __restrict__ vt) {
  __shared__ __align__(16) __hip_bfloat16 atile[16 * DIM];   // 24,576 B
  const int tid = threadIdx.x;
  const int lane = tid & 31;
  const int wave = tid >> 5;
  const int m0 = blockIdx.x * 16;
  const int c0 = blockIdx.y * 256 + wave * 32;

  {  // stage A tile (16 contiguous rows) via async copy
    const __hip_bfloat16* ag = xb + (size_t)m0 * DIM;
    for (int i = tid; i < (16 * DIM) / 8; i += 256)          // 1536 x 16B
      async_load_b128((unsigned)i * 16u, ag + (size_t)i * 8, atile);
    wait_asynccnt0();
  }
  __syncthreads();

  const __hip_bfloat16* brow0 = wb + (size_t)c0 * DIM;
  const __hip_bfloat16* brow1 = wb + (size_t)(c0 + 16) * DIM;
  v8f acc0 = {}, acc1 = {};
  for (int kk = 0; kk < DIM; kk += 32) {
    __builtin_prefetch(brow0 + (size_t)(lane & 15) * DIM + kk + 128, 0, 3);
    v16bf a = load_frag(atile, DIM, kk, lane);               // ds_load_b128 x2
    acc0 = wmma_bf16(a, load_frag(brow0, DIM, kk, lane), acc0);
    acc1 = wmma_bf16(a, load_frag(brow1, DIM, kk, lane), acc1);
  }
  const int hi8 = (lane & 16) ? 8 : 0;
  const int nlo = lane & 15;
#pragma unroll
  for (int t = 0; t < 2; ++t) {
    v8f acc = t ? acc1 : acc0;
#pragma unroll
    for (int r = 0; r < 8; ++r) {
      int m = m0 + r + hi8;                 // < 12608 always (788*16 exact)
      int c = c0 + t * 16 + nlo;
      int b = m / NTOK, tok = m % NTOK;
      int which = c / DIM, cc = c % DIM;
      int hh = cc >> 6, dd = cc & 63;
      int bh = b * NH + hh;
      float val = acc[r];
      if (which == 0) {
        val = (val + q_bias[cc]) * QSCALE;
        qs[((size_t)bh * NPAD + tok) * HD + dd] = __float2bfloat16(val);
      } else if (which == 1) {
        kmat[((size_t)bh * NPAD + tok) * HD + dd] = __float2bfloat16(val);
      } else {
        val += v_bias[cc];
        vt[((size_t)bh * HD + dd) * NPAD + tok] = __float2bfloat16(val);
      }
    }
  }
}

// ---------------- Attention: 1 block per (b,h), 13 waves ----------------
// LDS layout (single shared block, allocated at LDS offset 0):
//   [0,      28672) : V^T tile [64][224] bf16
//   [28672,  57344) : K  tile [224][64] bf16; after S-phase barrier this
//                     region is reused as per-wave P staging (13 * 1KB).
#define VL_OFF 0u
#define KL_OFF 28672u
#define TILE_BYTES 28672   // 224*64*2
#define TILE_CHUNKS 1792   // /16B

__global__ void __launch_bounds__(416) k_attn(
    const __hip_bfloat16* __restrict__ qs, const __hip_bfloat16* __restrict__ kmat,
    const __hip_bfloat16* __restrict__ vt, const float* __restrict__ rpb,
    const int* __restrict__ ridx, __hip_bfloat16* __restrict__ ao) {
  __shared__ __align__(16) unsigned char smem[2 * TILE_BYTES];
  const int tid = threadIdx.x;
  const int lane = tid & 31;
  const int wave = tid >> 5;              // 0..12 -> Q strip
  const int bh = blockIdx.x;
  const int b = bh / NH, h = bh % NH;
  const int hi8 = (lane & 16) ? 8 : 0;
  const int nlo = lane & 15;
  const int m0 = wave * 16;

  // Cooperative async copy of K and V^T tiles into LDS (ASYNCcnt-tracked).
  {
    const __hip_bfloat16* kg = kmat + (size_t)bh * NPAD * HD;
    const __hip_bfloat16* vg = vt + (size_t)bh * HD * NPAD;
    for (int i = tid; i < TILE_CHUNKS; i += 416) {
      async_load_b128(KL_OFF + (unsigned)i * 16u, kg + (size_t)i * 8, smem);
      async_load_b128(VL_OFF + (unsigned)i * 16u, vg + (size_t)i * 8, smem);
    }
    wait_asynccnt0();
  }
  __syncthreads();

  const __hip_bfloat16* vl = (const __hip_bfloat16*)(smem + VL_OFF);
  const __hip_bfloat16* kl = (const __hip_bfloat16*)(smem + KL_OFF);
  const __hip_bfloat16* qbase = qs + ((size_t)bh * NPAD + m0) * HD;

  // S = (Q*scale) @ K^T : 14 column tiles, K-dim = 64 (2 wmma each).
  // A-frag from global (wave-private rows), B-frag from LDS K tile.
  v8f sc[14];
#pragma unroll
  for (int j = 0; j < 14; ++j) {
    v8f acc = {};
#pragma unroll
    for (int ks = 0; ks < HD; ks += 32) {
      v16bf a  = load_frag(qbase, HD, ks, lane);
      v16bf bt = load_frag(kl + j * 16 * HD, HD, ks, lane);
      acc = wmma_bf16(a, bt, acc);
    }
    sc[j] = acc;
  }
  // relative position bias gather + pad mask
#pragma unroll
  for (int j = 0; j < 14; ++j) {
    int n = j * 16 + nlo;
#pragma unroll
    for (int r = 0; r < 8; ++r) {
      int m = m0 + r + hi8;
      if (m < NTOK && n < NTOK)
        sc[j][r] += rpb[ridx[m * NTOK + n] * NH + h];
      else
        sc[j][r] = -1e30f;
    }
  }
  // softmax: row m lives in one 16-lane half -> xor-shuffle reduce (masks 8..1)
#pragma unroll
  for (int r = 0; r < 8; ++r) {
    float mx = -1e30f;
#pragma unroll
    for (int j = 0; j < 14; ++j) mx = fmaxf(mx, sc[j][r]);
#pragma unroll
    for (int d = 8; d >= 1; d >>= 1) mx = fmaxf(mx, __shfl_xor(mx, d, 32));
    float sum = 0.f;
#pragma unroll
    for (int j = 0; j < 14; ++j) {
      float e = __expf(sc[j][r] - mx);
      sc[j][r] = e;
      sum += e;
    }
#pragma unroll
    for (int d = 8; d >= 1; d >>= 1) sum += __shfl_xor(sum, d, 32);
    float inv = 1.0f / sum;
#pragma unroll
    for (int j = 0; j < 14; ++j) sc[j][r] *= inv;
  }

  // All waves done reading the K tile -> reuse its LDS region for P staging.
  __syncthreads();
  __hip_bfloat16* myp =
      (__hip_bfloat16*)(smem + KL_OFF) + wave * (16 * 32);  // 1KB per wave

  // O = P @ V : relayout P (D-frag -> A-frag) via per-wave LDS tile; V^T rows
  // in LDS give contiguous B-frag reads. K-dim = 224 -> 7 steps of 32.
  v8f o0 = {}, o1 = {}, o2 = {}, o3 = {};
#pragma unroll
  for (int s7 = 0; s7 < 7; ++s7) {
#pragma unroll
    for (int t = 0; t < 2; ++t) {
      int j = 2 * s7 + t;
#pragma unroll
      for (int r = 0; r < 8; ++r)
        myp[(r + hi8) * 32 + t * 16 + nlo] = __float2bfloat16(sc[j][r]);
    }
    v16bf pa = load_frag(myp, 32, 0, lane);     // ds_load_b128 x2
    v16bf vb;
    vb = load_frag(vl + 0 * 16 * NPAD, NPAD, s7 * 32, lane); o0 = wmma_bf16(pa, vb, o0);
    vb = load_frag(vl + 1 * 16 * NPAD, NPAD, s7 * 32, lane); o1 = wmma_bf16(pa, vb, o1);
    vb = load_frag(vl + 2 * 16 * NPAD, NPAD, s7 * 32, lane); o2 = wmma_bf16(pa, vb, o2);
    vb = load_frag(vl + 3 * 16 * NPAD, NPAD, s7 * 32, lane); o3 = wmma_bf16(pa, vb, o3);
  }
  // store [B, tok, H*64] bf16 for proj GEMM
#pragma unroll
  for (int n2 = 0; n2 < 4; ++n2) {
    v8f o = (n2 == 0) ? o0 : (n2 == 1) ? o1 : (n2 == 2) ? o2 : o3;
#pragma unroll
    for (int r = 0; r < 8; ++r) {
      int m = m0 + r + hi8;
      if (m < NTOK)
        ao[((size_t)b * NTOK + m) * DIM + h * HD + n2 * 16 + nlo] =
            __float2bfloat16(o[r]);
    }
  }
}

// ---------------- Proj GEMM: [12608,768] @ W^T[768,768] + b -> fp32 ----------
__global__ void __launch_bounds__(256) k_proj(
    const __hip_bfloat16* __restrict__ ab, const __hip_bfloat16* __restrict__ wb,
    const float* __restrict__ bias, float* __restrict__ out) {
  __shared__ __align__(16) __hip_bfloat16 atile[16 * DIM];   // 24,576 B
  const int tid = threadIdx.x;
  const int lane = tid & 31;
  const int wave = tid >> 5;
  const int m0 = blockIdx.x * 16;
  const int c0 = blockIdx.y * 256 + wave * 32;

  {  // stage A tile via async copy
    const __hip_bfloat16* ag = ab + (size_t)m0 * DIM;
    for (int i = tid; i < (16 * DIM) / 8; i += 256)
      async_load_b128((unsigned)i * 16u, ag + (size_t)i * 8, atile);
    wait_asynccnt0();
  }
  __syncthreads();

  const __hip_bfloat16* brow0 = wb + (size_t)c0 * DIM;
  const __hip_bfloat16* brow1 = wb + (size_t)(c0 + 16) * DIM;
  v8f acc0 = {}, acc1 = {};
  for (int kk = 0; kk < DIM; kk += 32) {
    __builtin_prefetch(brow0 + (size_t)(lane & 15) * DIM + kk + 128, 0, 3);
    v16bf a = load_frag(atile, DIM, kk, lane);               // ds_load_b128 x2
    acc0 = wmma_bf16(a, load_frag(brow0, DIM, kk, lane), acc0);
    acc1 = wmma_bf16(a, load_frag(brow1, DIM, kk, lane), acc1);
  }
  const int hi8 = (lane & 16) ? 8 : 0;
  const int nlo = lane & 15;
#pragma unroll
  for (int t = 0; t < 2; ++t) {
    v8f acc = t ? acc1 : acc0;
    int c = c0 + t * 16 + nlo;
    float bv = bias[c];
#pragma unroll
    for (int r = 0; r < 8; ++r) {
      int m = m0 + r + hi8;
      out[(size_t)m * DIM + c] = acc[r] + bv;
    }
  }
}

extern "C" void kernel_launch(void* const* d_in, const int* in_sizes, int n_in,
                              void* d_out, int out_size, void* d_ws, size_t ws_size,
                              hipStream_t stream) {
  (void)in_sizes; (void)n_in; (void)out_size; (void)ws_size;
  const float* x      = (const float*)d_in[0];
  const float* qkv_w  = (const float*)d_in[1];
  const float* q_bias = (const float*)d_in[2];
  const float* v_bias = (const float*)d_in[3];
  const float* rpb    = (const float*)d_in[4];
  const float* proj_w = (const float*)d_in[5];
  const float* proj_b = (const float*)d_in[6];
  const int*   ridx   = (const int*)d_in[7];
  float* out = (float*)d_out;

  __hip_bfloat16* ws = (__hip_bfloat16*)d_ws;
  const size_t n_xb  = (size_t)MROWS * DIM;              // 9,682,944
  const size_t n_wq  = (size_t)3 * DIM * DIM;            // 1,769,472
  const size_t n_wp  = (size_t)DIM * DIM;                //   589,824
  const size_t n_qkv = (size_t)BATCH * NH * NPAD * HD;   // 11,010,048
  __hip_bfloat16* xb   = ws;
  __hip_bfloat16* wqb  = xb + n_xb;
  __hip_bfloat16* wpb  = wqb + n_wq;
  __hip_bfloat16* qs   = wpb + n_wp;
  __hip_bfloat16* kmat = qs + n_qkv;
  __hip_bfloat16* vt   = kmat + n_qkv;
  __hip_bfloat16* ao   = vt + n_qkv;                     // total ~104 MB

  k_convert<<<(int)((n_xb + 255) / 256), 256, 0, stream>>>(x, xb, (int)n_xb);
  k_convert<<<(int)((n_wq + 255) / 256), 256, 0, stream>>>(qkv_w, wqb, (int)n_wq);
  k_convert<<<(int)((n_wp + 255) / 256), 256, 0, stream>>>(proj_w, wpb, (int)n_wp);
  k_zero<<<(int)((n_qkv / 2 + 255) / 256), 256, 0, stream>>>((unsigned int*)vt,
                                                             (int)(n_qkv / 2));

  k_qkv<<<dim3(MTILES, 9), 256, 0, stream>>>(xb, wqb, q_bias, v_bias, qs, kmat, vt);
  k_attn<<<BATCH * NH, 416, 0, stream>>>(qs, kmat, vt, rpb, ridx, ao);
  k_proj<<<dim3(MTILES, 3), 256, 0, stream>>>(ao, wpb, proj_b, out);
}
